// DualPathAttention_47708496724246
// MI455X (gfx1250) — compile-verified, matrix operand order checked by reading, not run on hardware
//
#include <hip/hip_runtime.h>

// ---------------------------------------------------------------------------
// DualPathAttention for MI455X (gfx1250), wave32 + WMMA bf16-split (hi/lo)
// GEMM weight tiles staged to LDS via the Tensor Data Mover (TDM).
// ---------------------------------------------------------------------------
#define B_   8
#define N_   784
#define C_   512
#define H_   8
#define DH_  64
#define M_   (B_ * N_)      // 6272
#define MC_  (M_ * C_)      // 3211264

typedef unsigned short u16;
typedef __attribute__((ext_vector_type(16))) __bf16 v16bf;
typedef __attribute__((ext_vector_type(8)))  float  v8f;

union Frag16 {
    v16bf v;
    u16   u[16];
    uint4 q[2];
};

__device__ __forceinline__ v8f wmma_bf16(v16bf a, v16bf b, v8f c) {
    return __builtin_amdgcn_wmma_f32_16x16x32_bf16(
        /*neg_a=*/false, a, /*neg_b=*/false, b,
        /*c_mod=*/(short)0, c, /*reuse_a=*/false, /*reuse_b=*/false);
}

// bf16 round-to-nearest-even, pure bit math (no __bf16 arithmetic needed)
__device__ __forceinline__ u16 bf_rne(float f) {
    unsigned u = __builtin_bit_cast(unsigned, f);
    unsigned r = u + 0x7FFFu + ((u >> 16) & 1u);
    return (u16)(r >> 16);
}
__device__ __forceinline__ void split_f32(float f, u16& h, u16& l) {
    h = bf_rne(f);
    float fh = __builtin_bit_cast(float, (unsigned)h << 16);
    l = bf_rne(f - fh);
}

// Load one 16-bit WMMA operand fragment (A or B) from a row-major source.
// lane_base = pointer to this lane's row (row = lane%16 of the tile).
// Element e maps to k = k0 + (e<8 ? e : 8+e) + 8*(lane>=16)  [ISA 7.12.2].
__device__ __forceinline__ v16bf ld_frag(const u16* lane_base, int k0, int half) {
    Frag16 f;
    const u16* p = lane_base + k0 + half * 8;
    f.q[0] = *(const uint4*)(p);
    f.q[1] = *(const uint4*)(p + 16);
    return f.v;
}

// Row reductions across the 16-lane half that holds one C/D row's columns.
__device__ __forceinline__ float red_max16(float x) {
    x = fmaxf(x, __shfl_xor(x, 1, 32));
    x = fmaxf(x, __shfl_xor(x, 2, 32));
    x = fmaxf(x, __shfl_xor(x, 4, 32));
    x = fmaxf(x, __shfl_xor(x, 8, 32));
    return x;
}
__device__ __forceinline__ float red_sum16(float x) {
    x += __shfl_xor(x, 1, 32);
    x += __shfl_xor(x, 2, 32);
    x += __shfl_xor(x, 4, 32);
    x += __shfl_xor(x, 8, 32);
    return x;
}

// ---------------------------------------------------------------------------
// Tensor Data Mover: DMA a 2-D bf16 tile [tile_n lines x tile_k elems] from a
// row-major global matrix (line stride ld_elems) into contiguous LDS.
// D# layout per CDNA5 ISA sec.8 (group0 128b, group1 256b).
// This toolchain exposes the 6-arg builtin form:
//   (uint32x4 g0, int32x8 g1, int32x4, int32x4, int32x8, i32 cpol)
// 2-D tensor -> trailing groups zero-filled.
// ---------------------------------------------------------------------------
#if __has_builtin(__builtin_amdgcn_tensor_load_to_lds)
#define USE_TDM 1
typedef __attribute__((ext_vector_type(4))) unsigned int v4u_t;
typedef __attribute__((ext_vector_type(8))) int          v8i_t;
typedef __attribute__((ext_vector_type(4))) int          v4i_t;

__device__ __forceinline__ void tdm_load_tile_2d(unsigned lds_byte_addr,
                                                 const u16* gsrc,
                                                 unsigned tile_k, unsigned tile_n,
                                                 unsigned ld_elems) {
    unsigned long long ga = (unsigned long long)(size_t)gsrc;
    v4u_t g0 = {0u, 0u, 0u, 0u};
    g0[0] = 1u;                                   // count=1 (valid), user mode
    g0[1] = lds_byte_addr;                        // lds_addr        [63:32]
    g0[2] = (unsigned)ga;                         // global_addr lo  [95:64]
    g0[3] = (unsigned)((ga >> 32) & 0x01FFFFFFull)// global_addr hi  [120:96]
          | (2u << 30);                           // type=2 ("image")[127:126]
    v8i_t g1 = {0, 0, 0, 0, 0, 0, 0, 0};
    g1[0] = (int)(1u << 16);                      // wg_mask=0, data_size=1 (2B)
    g1[1] = (int)((ld_elems & 0xFFFFu) << 16);    // tensor_dim0[15:0]  @bit48
    g1[2] = (int)((ld_elems >> 16) | (8192u << 16)); // dim0[31:16], dim1[15:0]
    g1[3] = (int)(tile_k << 16);                  // dim1[31:16]=0, tile_dim0
    g1[4] = (int)tile_n;                          // tile_dim1 (tile_dim2 = 0)
    g1[5] = (int)ld_elems;                        // tensor_dim0_stride[31:0]
    g1[6] = 0;                                    // stride0[47:32], stride1 lo
    g1[7] = 0;                                    // stride1 hi
    v4i_t z4 = {0, 0, 0, 0};
    v8i_t z8 = {0, 0, 0, 0, 0, 0, 0, 0};
    __builtin_amdgcn_tensor_load_to_lds(g0, g1, z4, z4, z8, 0);
}
#else
#define USE_TDM 0
#endif

// ---------------------------------------------------------------------------
// fp32 -> bf16 hi/lo split, [rows,512] source, configurable dst stride/offset
// ---------------------------------------------------------------------------
__global__ void k_split(const float* __restrict__ src, u16* __restrict__ hi,
                        u16* __restrict__ lo, int total, int dst_stride, int col_off) {
    int i = blockIdx.x * 256 + threadIdx.x;
    if (i >= total) return;
    int r = i >> 9, c = i & 511;
    u16 h, l;
    split_f32(src[i], h, l);
    size_t o = (size_t)r * dst_stride + col_off + c;
    hi[o] = h; lo[o] = l;
}

// fp32 weight [K,512] -> transposed bf16 hi/lo [512,K] (contiguous B-frag rows)
__global__ void k_splitT(const float* __restrict__ src, u16* __restrict__ hi,
                         u16* __restrict__ lo, int K) {
    int i = blockIdx.x * 256 + threadIdx.x;   // total = K*512, exact
    int r = i >> 9, c = i & 511;
    u16 h, l;
    split_f32(src[i], h, l);
    size_t o = (size_t)c * K + r;
    hi[o] = h; lo[o] = l;
}

__global__ void k_zero16(u16* p, int n) {
    int i = blockIdx.x * 256 + threadIdx.x;
    if (i < n) p[i] = 0;
}

// ---------------------------------------------------------------------------
// Generic split-bf16 WMMA GEMM: out[M,512] = A[M,K] @ W[K,512] + bias
// A given as hi/lo bf16 (row-major, stride K); W given transposed [512,K].
// Block = 256 thr = 8 waves -> 128x64 tile; each wave 32x32 (2x2 subtiles).
// The 64x32 weight tile (hi+lo) is DMA'd to LDS by the TDM once per k-step;
// all 8 waves consume it as B fragments from LDS.
// mode 0: fp32 out; mode 1: bf16 hi/lo out; mode 2: bf16 hi/lo, per-batch
// transposed out_t[b][col][n] (for V of global attention).
// ---------------------------------------------------------------------------
__global__ __launch_bounds__(256) void k_gemm(
    const u16* __restrict__ Ah, const u16* __restrict__ Al,
    const u16* __restrict__ Wh, const u16* __restrict__ Wl,
    const float* __restrict__ bias,
    float* __restrict__ outf, u16* __restrict__ oh, u16* __restrict__ ol,
    int K, int mode) {
#if USE_TDM
    __shared__ u16 ldsBh[64 * 32];   // [n_local][k] bf16 hi plane, 4KB
    __shared__ u16 ldsBl[64 * 32];   // lo plane
#endif
    int tid  = threadIdx.x;
    int lane = tid & 31, l16 = lane & 15, half = lane >> 4;
    int w    = tid >> 5;
    int wr   = w & 3, wc = w >> 2;
    int m0   = blockIdx.x * 128 + wr * 32;
    int nblk = blockIdx.y * 64;
    int n0   = nblk + wc * 32;

    const u16 *ar[2], *arl[2];
#pragma unroll
    for (int s = 0; s < 2; ++s) {
        size_t a = (size_t)(m0 + s * 16 + l16) * K;
        ar[s] = Ah + a;  arl[s] = Al + a;
    }
#if !USE_TDM
    const u16 *br[2], *brl[2];
#pragma unroll
    for (int s = 0; s < 2; ++s) {
        size_t b = (size_t)(n0 + s * 16 + l16) * K;
        br[s] = Wh + b;  brl[s] = Wl + b;
    }
#endif
    v8f acc[2][2] = {};
    for (int k0 = 0; k0 < K; k0 += 32) {
#if USE_TDM
        if (w == 0) {
            // one DMA per plane for the whole workgroup's 64x32 weight tile
            tdm_load_tile_2d((unsigned)(size_t)ldsBh, Wh + (size_t)nblk * K + k0,
                             32u, 64u, (unsigned)K);
            tdm_load_tile_2d((unsigned)(size_t)ldsBl, Wl + (size_t)nblk * K + k0,
                             32u, 64u, (unsigned)K);
            __builtin_amdgcn_s_wait_tensorcnt(0);
        }
        __syncthreads();                       // tile visible to all waves
#endif
        if (k0 + 32 < K) {                     // prefetch next A k-tile
            __builtin_prefetch(ar[0] + k0 + 32, 0, 1);
            __builtin_prefetch(arl[0] + k0 + 32, 0, 1);
        }
        v16bf a_h[2], a_l[2], b_h[2], b_l[2];
#pragma unroll
        for (int s = 0; s < 2; ++s) {
            a_h[s] = ld_frag(ar[s],  k0, half);
            a_l[s] = ld_frag(arl[s], k0, half);
#if USE_TDM
            int nl = wc * 32 + s * 16 + l16;   // row within the 64-wide tile
            b_h[s] = ld_frag(&ldsBh[nl * 32], 0, half);
            b_l[s] = ld_frag(&ldsBl[nl * 32], 0, half);
#else
            b_h[s] = ld_frag(br[s],  k0, half);
            b_l[s] = ld_frag(brl[s], k0, half);
#endif
        }
#pragma unroll
        for (int sm = 0; sm < 2; ++sm)
#pragma unroll
            for (int sn = 0; sn < 2; ++sn) {
                acc[sm][sn] = wmma_bf16(a_h[sm], b_h[sn], acc[sm][sn]);
                acc[sm][sn] = wmma_bf16(a_h[sm], b_l[sn], acc[sm][sn]);
                acc[sm][sn] = wmma_bf16(a_l[sm], b_h[sn], acc[sm][sn]);
            }
#if USE_TDM
        __syncthreads();                       // reads done before next DMA
#endif
    }
#pragma unroll
    for (int sm = 0; sm < 2; ++sm)
#pragma unroll
        for (int sn = 0; sn < 2; ++sn) {
            int col  = n0 + sn * 16 + l16;
            float bv = bias[col];
#pragma unroll
            for (int v = 0; v < 8; ++v) {
                int   row = m0 + sm * 16 + v + half * 8;
                float val = acc[sm][sn][v] + bv;
                if (mode == 0) {
                    outf[(size_t)row * 512 + col] = val;
                } else if (mode == 1) {
                    u16 h, l; split_f32(val, h, l);
                    size_t o = (size_t)row * 512 + col;
                    oh[o] = h; ol[o] = l;
                } else {
                    int b = row / 784, n = row - b * 784;
                    u16 h, l; split_f32(val, h, l);
                    size_t o = ((size_t)b * 512 + col) * 784 + n;
                    oh[o] = h; ol[o] = l;
                }
            }
        }
}

// ---------------------------------------------------------------------------
// Global attention, flash-style: 1 wave per (b, head, 16-row q-tile).
// 64 bh-problems * 49 tiles = 3136 waves; 25 key blocks of 32 (last masked).
// V supplied per-batch transposed: Vt[b][col][n] so PV B-frags are contiguous.
// P is re-laid out (C-layout -> A-layout) through per-wave LDS.
// ---------------------------------------------------------------------------
__global__ __launch_bounds__(128) void k_attn_g(
    const u16* __restrict__ Qh, const u16* __restrict__ Ql,
    const u16* __restrict__ Kh, const u16* __restrict__ Kl,
    const u16* __restrict__ Vh, const u16* __restrict__ Vl,
    float* __restrict__ out) {
    __shared__ u16 lph[4][512];
    __shared__ u16 lpl[4][512];
    int tid  = threadIdx.x;
    int lane = tid & 31, l16 = lane & 15, half = lane >> 4;
    int w    = tid >> 5;
    int gid  = blockIdx.x * 4 + w;
    int bh   = gid / 49, qt = gid % 49;
    int b    = bh >> 3, h = bh & 7;
    int ch   = h * 64;

    size_t rq = (size_t)(b * 784 + qt * 16 + l16) * 512 + ch;
    v16bf qh_[2], ql_[2];
#pragma unroll
    for (int kt = 0; kt < 2; ++kt) {
        qh_[kt] = ld_frag(Qh + rq, kt * 32, half);
        ql_[kt] = ld_frag(Ql + rq, kt * 32, half);
    }
    v8f   acc[4] = {};
    float m_i[8], l_i[8];
#pragma unroll
    for (int v = 0; v < 8; ++v) { m_i[v] = -3e38f; l_i[v] = 0.f; }

    for (int blk = 0; blk < 25; ++blk) {
        int j0 = blk * 32;
        v8f s[2] = {};
#pragma unroll
        for (int sn = 0; sn < 2; ++sn) {
            int key = j0 + sn * 16 + l16;
            int kc  = key > 783 ? 783 : key;
            size_t rk = (size_t)(b * 784 + kc) * 512 + ch;
#pragma unroll
            for (int kt = 0; kt < 2; ++kt) {
                v16bf kbh = ld_frag(Kh + rk, kt * 32, half);
                v16bf kbl = ld_frag(Kl + rk, kt * 32, half);
                s[sn] = wmma_bf16(qh_[kt], kbh, s[sn]);
                s[sn] = wmma_bf16(qh_[kt], kbl, s[sn]);
                s[sn] = wmma_bf16(ql_[kt], kbh, s[sn]);
            }
        }
        bool ok0 = (j0 + l16) < 784;
        bool ok1 = (j0 + 16 + l16) < 784;
#pragma unroll
        for (int v = 0; v < 8; ++v) {
            float s0 = ok0 ? s[0][v] * 0.125f : -1e30f;
            float s1 = ok1 ? s[1][v] * 0.125f : -1e30f;
            float mx = red_max16(fmaxf(s0, s1));
            float nm = fmaxf(m_i[v], mx);
            float cr = __expf(m_i[v] - nm);
            m_i[v]   = nm;
            float p0 = __expf(s0 - nm);
            float p1 = __expf(s1 - nm);
            l_i[v]   = l_i[v] * cr + red_sum16(p0 + p1);
#pragma unroll
            for (int t = 0; t < 4; ++t) acc[t][v] *= cr;
            int row = v + half * 8;
            u16 h0, l0, h1, l1;
            split_f32(p0, h0, l0);
            split_f32(p1, h1, l1);
            lph[w][row * 32 + l16]      = h0;  lpl[w][row * 32 + l16]      = l0;
            lph[w][row * 32 + 16 + l16] = h1;  lpl[w][row * 32 + 16 + l16] = l1;
        }
        asm volatile("s_wait_dscnt 0" ::: "memory");  // LDS RAW within wave
        v16bf pah = ld_frag(&lph[w][(size_t)l16 * 32], 0, half);
        v16bf pal = ld_frag(&lpl[w][(size_t)l16 * 32], 0, half);
#pragma unroll
        for (int t = 0; t < 4; ++t) {
            int col = ch + t * 16 + l16;
            size_t rv = ((size_t)b * 512 + col) * 784 + j0;
            v16bf vbh = ld_frag(Vh + rv, 0, half);
            v16bf vbl = ld_frag(Vl + rv, 0, half);
            acc[t] = wmma_bf16(pah, vbh, acc[t]);
            acc[t] = wmma_bf16(pah, vbl, acc[t]);
            acc[t] = wmma_bf16(pal, vbh, acc[t]);
        }
    }
#pragma unroll
    for (int t = 0; t < 4; ++t) {
        int col = ch + t * 16 + l16;
#pragma unroll
        for (int v = 0; v < 8; ++v) {
            int row = b * 784 + qt * 16 + v + half * 8;
            out[(size_t)row * 512 + col] = acc[t][v] / l_i[v];
        }
    }
}

// ---------------------------------------------------------------------------
// Local (7x7 window) attention: 1 wave per (window, head, 16-row q-tile).
// 49 tokens -> padded to 64 keys (masked), full softmax (no streaming needed).
// ---------------------------------------------------------------------------
__global__ __launch_bounds__(128) void k_attn_l(
    const u16* __restrict__ Qh, const u16* __restrict__ Ql,
    const u16* __restrict__ Kh, const u16* __restrict__ Kl,
    const u16* __restrict__ Vh, const u16* __restrict__ Vl,
    float* __restrict__ out) {
    __shared__ u16 lph[4][1024];
    __shared__ u16 lpl[4][1024];
    int tid  = threadIdx.x;
    int lane = tid & 31, l16 = lane & 15, half = lane >> 4;
    int w    = tid >> 5;
    int gid  = blockIdx.x * 4 + w;
    int win  = gid >> 5;
    int rest = gid & 31;
    int h    = rest >> 2, qt = rest & 3;
    int b    = win >> 4;
    int ww   = win & 15;
    int ih   = ww >> 2, iw = ww & 3;
    int ch   = h * 64;
    int rowbase = b * 784 + ih * 7 * 28 + iw * 7;
    auto row_of = [&](int tok) {
        int r = tok / 7, c = tok - r * 7;
        return rowbase + r * 28 + c;
    };

    int qtok = qt * 16 + l16;
    int qc   = qtok > 48 ? 48 : qtok;
    size_t rq = (size_t)row_of(qc) * 512 + ch;
    v16bf qh_[2], ql_[2];
#pragma unroll
    for (int kt = 0; kt < 2; ++kt) {
        qh_[kt] = ld_frag(Qh + rq, kt * 32, half);
        ql_[kt] = ld_frag(Ql + rq, kt * 32, half);
    }
    v8f s[4] = {};
#pragma unroll
    for (int sn = 0; sn < 4; ++sn) {
        int ktok = sn * 16 + l16;
        int kc   = ktok > 48 ? 48 : ktok;
        size_t rk = (size_t)row_of(kc) * 512 + ch;
#pragma unroll
        for (int kt = 0; kt < 2; ++kt) {
            v16bf kbh = ld_frag(Kh + rk, kt * 32, half);
            v16bf kbl = ld_frag(Kl + rk, kt * 32, half);
            s[sn] = wmma_bf16(qh_[kt], kbh, s[sn]);
            s[sn] = wmma_bf16(qh_[kt], kbl, s[sn]);
            s[sn] = wmma_bf16(ql_[kt], kbh, s[sn]);
        }
    }
    float l_i[8];
#pragma unroll
    for (int v = 0; v < 8; ++v) {
        float sv[4], mx = -3e38f;
#pragma unroll
        for (int t = 0; t < 4; ++t) {
            int key = t * 16 + l16;
            sv[t] = (key < 49) ? s[t][v] * 0.125f : -1e30f;
            mx = fmaxf(mx, sv[t]);
        }
        mx = red_max16(mx);
        float sum = 0.f;
        int   row = v + half * 8;
#pragma unroll
        for (int t = 0; t < 4; ++t) {
            float p = __expf(sv[t] - mx);
            sum += p;
            u16 hh, ll;
            split_f32(p, hh, ll);
            lph[w][row * 64 + t * 16 + l16] = hh;
            lpl[w][row * 64 + t * 16 + l16] = ll;
        }
        l_i[v] = red_sum16(sum);
    }
    asm volatile("s_wait_dscnt 0" ::: "memory");
    v8f acc[4] = {};
#pragma unroll
    for (int kt = 0; kt < 2; ++kt) {
        v16bf pah = ld_frag(&lph[w][(size_t)l16 * 64], kt * 32, half);
        v16bf pal = ld_frag(&lpl[w][(size_t)l16 * 64], kt * 32, half);
#pragma unroll
        for (int t = 0; t < 4; ++t) {
            int col = ch + t * 16 + l16;
            Frag16 fh, fl;
#pragma unroll
            for (int e = 0; e < 16; ++e) {
                int k   = (e < 8 ? e : 8 + e) + half * 8 + kt * 32;
                int tok = k > 48 ? 48 : k;
                size_t a = (size_t)row_of(tok) * 512 + col;
                fh.u[e] = Vh[a];
                fl.u[e] = Vl[a];
            }
            acc[t] = wmma_bf16(pah, fh.v, acc[t]);
            acc[t] = wmma_bf16(pah, fl.v, acc[t]);
            acc[t] = wmma_bf16(pal, fh.v, acc[t]);
        }
    }
#pragma unroll
    for (int t = 0; t < 4; ++t) {
        int col = ch + t * 16 + l16;
#pragma unroll
        for (int v = 0; v < 8; ++v) {
            int tok = qt * 16 + v + half * 8;
            if (tok < 49)
                out[(size_t)row_of(tok) * 512 + col] = acc[t][v] / l_i[v];
        }
    }
}

// ---------------------------------------------------------------------------
// Gated fusion: out = sigmoid(g)*yg + (1-sigmoid(g))*yli
// ---------------------------------------------------------------------------
__global__ void k_blend(const float* __restrict__ g, const float* __restrict__ yg,
                        const float* __restrict__ yli, float* __restrict__ out, int total) {
    int i = blockIdx.x * 256 + threadIdx.x;
    if (i >= total) return;
    float a = 1.f / (1.f + __expf(-g[i]));
    out[i] = a * yg[i] + (1.f - a) * yli[i];
}

// ---------------------------------------------------------------------------
extern "C" void kernel_launch(void* const* d_in, const int* in_sizes, int n_in,
                              void* d_out, int out_size, void* d_ws, size_t ws_size,
                              hipStream_t stream) {
    (void)in_sizes; (void)n_in; (void)out_size; (void)ws_size;
    const float* x   = (const float*)d_in[0];
    const float* bqg = (const float*)d_in[2];
    const float* bkg = (const float*)d_in[4];
    const float* bvg = (const float*)d_in[6];
    const float* bql = (const float*)d_in[8];
    const float* bkl = (const float*)d_in[10];
    const float* bvl = (const float*)d_in[12];
    const float* bi  = (const float*)d_in[14];
    const float* bg  = (const float*)d_in[16];
    const float* wsrc[8] = { (const float*)d_in[1], (const float*)d_in[3],
                             (const float*)d_in[5], (const float*)d_in[7],
                             (const float*)d_in[9], (const float*)d_in[11],
                             (const float*)d_in[13], (const float*)d_in[15] };
    const int wK[8] = { 512, 512, 512, 512, 512, 512, 512, 1024 };

    // ---- workspace bump allocator (all chunk sizes are 256B multiples) ----
    size_t off = 0;
    auto alloc = [&](size_t bytes) -> char* {
        off = (off + 255) & ~(size_t)255;
        char* p = (char*)d_ws + off;
        off += bytes;
        return p;
    };
    const size_t MCe  = (size_t)MC_;
    const size_t PADB = (MCe + 256) * 2;  // u16 buffer + tail pad

    u16* xh = (u16*)alloc(PADB);  u16* xl = (u16*)alloc(PADB);
    u16 *wt_h[8], *wt_l[8];
    for (int i = 0; i < 8; ++i) {
        wt_h[i] = (u16*)alloc((size_t)wK[i] * 512 * 2);
        wt_l[i] = (u16*)alloc((size_t)wK[i] * 512 * 2);
    }
    u16* qgh = (u16*)alloc(PADB); u16* qgl = (u16*)alloc(PADB);
    u16* kgh = (u16*)alloc(PADB); u16* kgl = (u16*)alloc(PADB);
    u16* vgh = (u16*)alloc(PADB); u16* vgl = (u16*)alloc(PADB);
    u16* qlh = (u16*)alloc(PADB); u16* qll = (u16*)alloc(PADB);
    u16* klh = (u16*)alloc(PADB); u16* kll = (u16*)alloc(PADB);
    u16* vlh = (u16*)alloc(PADB); u16* vll = (u16*)alloc(PADB);
    float* yg = (float*)alloc(MCe * 4);
    float* yl = (float*)alloc(MCe * 4);
    // --- aliases into dead buffers (safe: chunk sizes are 256B multiples) ---
    u16*   ylh  = xh;            // x dead after the 6 QKV GEMMs
    u16*   yll  = xl;
    float* yli  = (float*)qlh;   // ql dead after local attention (spans qlh+qll)
    u16*   cath = klh;           // kl dead after local attention (spans klh+kll)
    u16*   catl = kgh;           // kg dead after global attention (spans kgh+kgl)
    float* gbuf = yl;            // yl dead after its hi/lo split

    const int SPLIT_GRID = MC_ / 256;  // 12544
    dim3 gg(49, 8);                    // 6272/128 x 512/64

    // 1) precision splits
    k_split <<<SPLIT_GRID, 256, 0, stream>>>(x, xh, xl, MC_, 512, 0);
    for (int i = 0; i < 8; ++i)
        k_splitT<<<wK[i] * 512 / 256, 256, 0, stream>>>(wsrc[i], wt_h[i], wt_l[i], wK[i]);

    // 2) QKV projections (global + local), WMMA split-bf16
    k_gemm<<<gg, 256, 0, stream>>>(xh, xl, wt_h[0], wt_l[0], bqg, nullptr, qgh, qgl, 512, 1);
    k_gemm<<<gg, 256, 0, stream>>>(xh, xl, wt_h[1], wt_l[1], bkg, nullptr, kgh, kgl, 512, 1);
    k_gemm<<<gg, 256, 0, stream>>>(xh, xl, wt_h[2], wt_l[2], bvg, nullptr, vgh, vgl, 512, 2);
    k_gemm<<<gg, 256, 0, stream>>>(xh, xl, wt_h[3], wt_l[3], bql, nullptr, qlh, qll, 512, 1);
    k_gemm<<<gg, 256, 0, stream>>>(xh, xl, wt_h[4], wt_l[4], bkl, nullptr, klh, kll, 512, 1);
    k_gemm<<<gg, 256, 0, stream>>>(xh, xl, wt_h[5], wt_l[5], bvl, nullptr, vlh, vll, 512, 1);
    k_zero16<<<1, 256, 0, stream>>>(vgh + MCe, 256);  // tail pad for masked Vt reads
    k_zero16<<<1, 256, 0, stream>>>(vgl + MCe, 256);

    // 3) attention
    k_attn_g<<<784, 128, 0, stream>>>(qgh, qgl, kgh, kgl, vgh, vgl, yg);
    k_attn_l<<<1024, 128, 0, stream>>>(qlh, qll, klh, kll, vlh, vll, yl);

    // 4) y_li = y_local @ wi + bi
    k_split<<<SPLIT_GRID, 256, 0, stream>>>(yl, ylh, yll, MC_, 512, 0);
    k_gemm <<<gg, 256, 0, stream>>>(ylh, yll, wt_h[6], wt_l[6], bi, yli, nullptr, nullptr, 512, 0);

    // 5) gate: g = concat(y_global, y_li) @ wg + bg ; out = blend
    k_split<<<SPLIT_GRID, 256, 0, stream>>>(yg,  cath, catl, MC_, 1024, 0);
    k_split<<<SPLIT_GRID, 256, 0, stream>>>(yli, cath, catl, MC_, 1024, 512);
    k_gemm <<<gg, 256, 0, stream>>>(cath, catl, wt_h[7], wt_l[7], bg, gbuf, nullptr, nullptr, 1024, 0);
    k_blend<<<SPLIT_GRID, 256, 0, stream>>>(gbuf, yg, yli, (float*)d_out, MC_);
}
